// GroupedQueryAttention_37185826849448
// MI455X (gfx1250) — matrix operand order repaired and checked by reading, not compiled, not run on hardware
//
#include <hip/hip_runtime.h>
#include <hip/hip_bf16.h>

typedef __attribute__((ext_vector_type(16))) __bf16 v16bf;
typedef __attribute__((ext_vector_type(8)))  float  v8f;

__device__ __forceinline__ v8f wmma_bf16f32(v16bf a, v16bf b, v8f c) {
  // emits v_wmma_f32_16x16x32_bf16
  return __builtin_amdgcn_wmma_f32_16x16x32_bf16(false, a, false, b, (short)0, c, false, false);
}

// ---------------------------------------------------------------------------
// Pack fp32 row-major (rows x cols, leading dim ld) into WMMA A-fragments.
// frag index = mt*(cols/32)+kb. Storage: frag*512 + lane*16 + e (bf16 units)
// A 16x32 layout: lane m = lane&15, hi = lane>>4, K(e) = e<8 ? e+8*hi : e+8+8*hi
// ---------------------------------------------------------------------------
__global__ void pack_a_frag(const float* __restrict__ src, int ld, int rows, int cols,
                            __bf16* __restrict__ dst) {
  int tid  = blockIdx.x * blockDim.x + threadIdx.x;
  int lane = tid & 31;
  int frag = tid >> 5;
  int kbn  = cols >> 5;
  int nf   = (rows >> 4) * kbn;
  if (frag >= nf) return;
  int mt = frag / kbn, kb = frag - mt * kbn;
  int m  = mt * 16 + (lane & 15);
  int hi = lane >> 4;
  const float* s = src + (size_t)m * ld + kb * 32 + hi * 8;
  __bf16* o = dst + (size_t)frag * 512 + lane * 16;
#pragma unroll
  for (int e = 0; e < 16; ++e) {
    int k = (e < 8) ? e : (e + 8);
    o[e] = (__bf16)s[k];
  }
}

// ---------------------------------------------------------------------------
// Pack W (Nn x Kk row-major) as B-fragments of W^T: B[k][n] = W[n][k].
// frag = nt*(Kk/32)+kb. n = lane&15, hi = lane>>4, K(e) = e<8 ? e+8*hi : e+8+8*hi.
// ---------------------------------------------------------------------------
__global__ void pack_b_t(const float* __restrict__ W, int Nn, int Kk,
                         __bf16* __restrict__ dst) {
  int tid  = blockIdx.x * blockDim.x + threadIdx.x;
  int lane = tid & 31;
  int frag = tid >> 5;
  int kbn  = Kk >> 5;
  int nf   = (Nn >> 4) * kbn;
  if (frag >= nf) return;
  int nt = frag / kbn, kb = frag - nt * kbn;
  const float* s = W + (size_t)(nt * 16 + (lane & 15)) * Kk + kb * 32 + (lane >> 4) * 8;
  __bf16* o = dst + (size_t)frag * 512 + lane * 16;
#pragma unroll
  for (int e = 0; e < 16; ++e) o[e] = (__bf16)s[(e < 8) ? e : (e + 8)];
}

// ---------------------------------------------------------------------------
// Pack V (Krows x Nn row-major) as non-transposed B-fragments for P@V.
// frag = kt*(Nn/16)+dt; col = lane&15, hi = lane>>4; elem e<8 = key
// kt*16 + 8*hi + e; elems 8..15 zero (matched by zero-padded A).
// ---------------------------------------------------------------------------
__global__ void pack_b_n(const float* __restrict__ V, int Krows, int Nn,
                         __bf16* __restrict__ dst) {
  int tid  = blockIdx.x * blockDim.x + threadIdx.x;
  int lane = tid & 31;
  int frag = tid >> 5;
  int ntn  = Nn >> 4;
  int nf   = (Krows >> 4) * ntn;
  if (frag >= nf) return;
  int kt = frag / ntn, dt = frag - kt * ntn;
  int key = kt * 16 + (lane >> 4) * 8;
  __bf16* o = dst + (size_t)frag * 512 + lane * 16;
  if (key < Krows) {
    const float* s = V + (size_t)key * Nn + dt * 16 + (lane & 15);
#pragma unroll
    for (int e = 0; e < 8; ++e) o[e] = (__bf16)s[(size_t)e * Nn];
  } else {
#pragma unroll
    for (int e = 0; e < 8; ++e) o[e] = (__bf16)0.0f;
  }
#pragma unroll
  for (int e = 8; e < 16; ++e) o[e] = (__bf16)0.0f;
}

// ---------------------------------------------------------------------------
// GEMM: C(MxN) = A(MxK) * B(KxN) from pre-packed bf16 fragments.
// 8 waves / block, each wave owns a 16x64 tile (4 WMMAs per 32-wide K step).
// ---------------------------------------------------------------------------
__global__ void gemm_wmma_bf16(const v16bf* __restrict__ Ap, const v16bf* __restrict__ Bp,
                               float* __restrict__ Cm, int M, int N, int K) {
  (void)M;
  int lane = threadIdx.x & 31;
  int wv   = threadIdx.x >> 5;
  int kbn  = K >> 5;
  int mt   = blockIdx.y * 8 + wv;
  int ntb  = blockIdx.x * 4;
  const v16bf* A  = Ap + ((size_t)mt * kbn) * 32 + lane;
  const v16bf* B0 = Bp + ((size_t)(ntb + 0) * kbn) * 32 + lane;
  const v16bf* B1 = Bp + ((size_t)(ntb + 1) * kbn) * 32 + lane;
  const v16bf* B2 = Bp + ((size_t)(ntb + 2) * kbn) * 32 + lane;
  const v16bf* B3 = Bp + ((size_t)(ntb + 3) * kbn) * 32 + lane;
  v8f a0 = {}, a1 = {}, a2 = {}, a3 = {};
  for (int kb = 0; kb < kbn; ++kb) {
    v16bf av = A[(size_t)kb * 32];
    if (kb + 1 < kbn) __builtin_prefetch(&A[(size_t)(kb + 1) * 32], 0, 0);
    a0 = wmma_bf16f32(av, B0[(size_t)kb * 32], a0);
    a1 = wmma_bf16f32(av, B1[(size_t)kb * 32], a1);
    a2 = wmma_bf16f32(av, B2[(size_t)kb * 32], a2);
    a3 = wmma_bf16f32(av, B3[(size_t)kb * 32], a3);
  }
  int hf = lane >> 4, cn = lane & 15;
  size_t rb = (size_t)mt * 16 + hf * 8;
#pragma unroll
  for (int r = 0; r < 8; ++r) {
    float* row = Cm + (rb + r) * N + cn;
    row[(ntb + 0) * 16] = a0[r];
    row[(ntb + 1) * 16] = a1[r];
    row[(ntb + 2) * 16] = a2[r];
    row[(ntb + 3) * 16] = a3[r];
  }
}

// ---------------------------------------------------------------------------
// Fused RMSNorm + interleaved RoPE on q (4 heads x 256) and k (256), in place.
// One block per token row; waves 0-3 -> q heads, wave 4 -> k.
// ---------------------------------------------------------------------------
__global__ void rmsnorm_rope_qk(float* __restrict__ q, float* __restrict__ k,
                                const float* __restrict__ qg, const float* __restrict__ kg,
                                int T) {
  int row  = blockIdx.x;     // 0 .. B*T-1
  int t    = row & (T - 1);  // T is a power of two
  int wv   = threadIdx.x >> 5;
  int lane = threadIdx.x & 31;
  if (wv >= 5) return;
  float* p; const float* g;
  if (wv < 4) { p = q + (size_t)row * 1024 + wv * 256; g = qg; }
  else        { p = k + (size_t)row * 256;             g = kg; }
  float v[8], ss = 0.f;
#pragma unroll
  for (int j = 0; j < 8; ++j) { v[j] = p[lane * 8 + j]; ss += v[j] * v[j]; }
#pragma unroll
  for (int m = 1; m < 32; m <<= 1) ss += __shfl_xor(ss, m, 32);
  float rn = rsqrtf(ss * (1.0f / 256.0f) + 1e-6f);
#pragma unroll
  for (int j = 0; j < 4; ++j) {
    int d0 = lane * 8 + 2 * j;
    float x0 = v[2 * j] * rn * g[d0];
    float x1 = v[2 * j + 1] * rn * g[d0 + 1];
    int pr = lane * 4 + j;                       // rotary pair index 0..127
    float inv = __expf(-(float)pr * (13.815510558f / 128.0f));  // 1e6^(-p/128)
    float ang = (float)t * inv;
    float c, s;
    __sincosf(ang, &c, &s);
    p[d0]     = x0 * c - x1 * s;
    p[d0 + 1] = x0 * s + x1 * c;
  }
}

// ---------------------------------------------------------------------------
// Sliding-window causal attention, flash-style online softmax.
// One wave per (b, h, 16-query tile); scores + PV both via WMMA bf16.
// ---------------------------------------------------------------------------
__global__ void attn_swa_wmma(const v16bf* __restrict__ Qp, const v16bf* __restrict__ Kp,
                              const v16bf* __restrict__ Vp, float* __restrict__ out, int T) {
  __shared__ float plds[8][16][16];
  int lane = threadIdx.x & 31;
  int wv   = threadIdx.x >> 5;
  int nqt  = T >> 4;
  int task = blockIdx.x * 8 + wv;
  int qt = task % nqt;
  int bh = task / nqt;
  int h  = bh & 3, b = bh >> 2;
  int hf = lane >> 4, cn = lane & 15;

  const v16bf* Qf = Qp + (((size_t)(b * 4 + h) * nqt + qt) * 8) * 32 + lane;
  v16bf qf[8];
#pragma unroll
  for (int kb = 0; kb < 8; ++kb) qf[kb] = Qf[(size_t)kb * 32];

  const v16bf* Kb = Kp + ((size_t)b * nqt * 8) * 32 + lane;
  const v16bf* Vb = Vp + ((size_t)b * nqt * 16) * 32 + lane;

  v8f zero = {};
  v8f acc[16];
#pragma unroll
  for (int dt = 0; dt < 16; ++dt) acc[dt] = zero;
  float mrow[8], lrow[8];
#pragma unroll
  for (int r = 0; r < 8; ++r) { mrow[r] = -__builtin_inff(); lrow[r] = 0.f; }

  int ktlo = qt - 32; if (ktlo < 0) ktlo = 0;
  for (int kt = ktlo; kt <= qt; ++kt) {
    // scores S = Q * K^T for this 16-key tile (K-dim = 256 -> 8 WMMAs)
    v8f s = zero;
    const v16bf* Kf = Kb + (size_t)(kt * 8) * 32;
#pragma unroll
    for (int kb = 0; kb < 8; ++kb) s = wmma_bf16f32(qf[kb], Kf[(size_t)kb * 32], s);

    int j = kt * 16 + cn;
    float corr[8];
#pragma unroll
    for (int r = 0; r < 8; ++r) {
      int i = qt * 16 + hf * 8 + r;
      float sv = s[r] * 0.0625f;                 // 1/sqrt(256)
      if (j > i || j + 512 < i) sv = -__builtin_inff();
      float mx = sv;
#pragma unroll
      for (int mm = 1; mm < 16; mm <<= 1) mx = fmaxf(mx, __shfl_xor(mx, mm, 32));
      float mn = fmaxf(mrow[r], mx);
      float c  = __expf(mrow[r] - mn);
      float pv = __expf(sv - mn);
      float ps = pv;
#pragma unroll
      for (int mm = 1; mm < 16; mm <<= 1) ps += __shfl_xor(ps, mm, 32);
      lrow[r] = lrow[r] * c + ps;
      mrow[r] = mn;
      corr[r] = c;
      plds[wv][hf * 8 + r][cn] = pv;             // stage P for layout change
    }
    asm volatile("s_wait_dscnt 0x0" ::: "memory"); // intra-wave LDS RAW guard
#pragma unroll
    for (int dt = 0; dt < 16; ++dt)
#pragma unroll
      for (int r = 0; r < 8; ++r) acc[dt][r] *= corr[r];

    // rebuild P as a WMMA A-fragment (K = 16 real keys, upper half zero)
    v16bf af;
#pragma unroll
    for (int e = 0; e < 8; ++e) af[e] = (__bf16)plds[wv][cn][hf * 8 + e];
#pragma unroll
    for (int e = 8; e < 16; ++e) af[e] = (__bf16)0.0f;

    const v16bf* Vf = Vb + (size_t)(kt * 16) * 32;
#pragma unroll
    for (int dt = 0; dt < 16; ++dt) acc[dt] = wmma_bf16f32(af, Vf[(size_t)dt * 32], acc[dt]);
  }

  size_t rowb = (size_t)b * T + qt * 16 + hf * 8;
#pragma unroll
  for (int dt = 0; dt < 16; ++dt)
#pragma unroll
    for (int r = 0; r < 8; ++r)
      out[(rowb + r) * 1024 + h * 256 + dt * 16 + cn] = acc[dt][r] / lrow[r];
}

// ---------------------------------------------------------------------------
extern "C" void kernel_launch(void* const* d_in, const int* in_sizes, int n_in,
                              void* d_out, int out_size, void* d_ws, size_t ws_size,
                              hipStream_t stream) {
  (void)in_sizes; (void)n_in; (void)out_size; (void)ws_size;
  const float* x  = (const float*)d_in[0];
  const float* Wq = (const float*)d_in[1];
  const float* Wk = (const float*)d_in[2];
  const float* Wv = (const float*)d_in[3];
  const float* Wo = (const float*)d_in[4];
  const float* qg = (const float*)d_in[5];
  const float* kg = (const float*)d_in[6];
  float* out = (float*)d_out;

  const int B = 2, T = 4096, C = 1024, H = 4, HD = 256;
  const int R   = B * T;   // 8192 activation rows
  const int nqt = T / 16;  // 256 query tiles per (b,h)

  char* ws = (char*)d_ws;
  size_t off = 0;
  auto alloc = [&](size_t bytes) {
    char* p = ws + off;
    off += (bytes + 255) & ~(size_t)255;
    return p;
  };
  __bf16* Xp  = (__bf16*)alloc((size_t)(R / 16) * (C / 32) * 1024);   // 16 MB (reused)
  __bf16* Wqp = (__bf16*)alloc((size_t)(C / 16) * (C / 32) * 1024);   // 2 MB
  __bf16* Wkp = (__bf16*)alloc((size_t)(HD / 16) * (C / 32) * 1024);  // 0.5 MB
  __bf16* Wvp = (__bf16*)alloc((size_t)(HD / 16) * (C / 32) * 1024);  // 0.5 MB
  __bf16* Wop = (__bf16*)alloc((size_t)(C / 16) * (C / 32) * 1024);   // 2 MB
  float* qbuf = (float*)alloc((size_t)R * C * 4);                     // 32 MB (also attn_out)
  float* kbuf = (float*)alloc((size_t)R * HD * 4);                    // 8 MB
  float* vbuf = (float*)alloc((size_t)R * HD * 4);                    // 8 MB
  __bf16* Qp  = (__bf16*)alloc((size_t)B * H * nqt * 8 * 1024);       // 16 MB
  __bf16* Kp  = (__bf16*)alloc((size_t)B * nqt * 8 * 1024);           // 4 MB
  __bf16* Vp  = (__bf16*)alloc((size_t)B * nqt * 16 * 1024);          // 8 MB

  auto packBlocks = [](int nf) { return (nf * 32 + 255) / 256; };

  // 1) pack x into A-fragments
  { int nf = (R / 16) * (C / 32);
    pack_a_frag<<<packBlocks(nf), 256, 0, stream>>>(x, C, R, C, Xp); }
  // 2) pack weights (as B-fragments of W^T)
  { int nf = (C / 16) * (C / 32);
    pack_b_t<<<packBlocks(nf), 256, 0, stream>>>(Wq, C, C, Wqp);
    pack_b_t<<<packBlocks(nf), 256, 0, stream>>>(Wo, C, C, Wop); }
  { int nf = (HD / 16) * (C / 32);
    pack_b_t<<<packBlocks(nf), 256, 0, stream>>>(Wk, HD, C, Wkp);
    pack_b_t<<<packBlocks(nf), 256, 0, stream>>>(Wv, HD, C, Wvp); }
  // 3) q/k/v projections (WMMA)
  gemm_wmma_bf16<<<dim3(C / 64, R / 128), 256, 0, stream>>>(
      (const v16bf*)Xp, (const v16bf*)Wqp, qbuf, R, C, C);
  gemm_wmma_bf16<<<dim3(HD / 64, R / 128), 256, 0, stream>>>(
      (const v16bf*)Xp, (const v16bf*)Wkp, kbuf, R, HD, C);
  gemm_wmma_bf16<<<dim3(HD / 64, R / 128), 256, 0, stream>>>(
      (const v16bf*)Xp, (const v16bf*)Wvp, vbuf, R, HD, C);
  // 4) fused RMSNorm + RoPE (in place)
  rmsnorm_rope_qk<<<R, 256, 0, stream>>>(qbuf, kbuf, qg, kg, T);
  // 5) pack q/k/v into attention fragment layouts
  for (int b = 0; b < B; ++b) {
    for (int h = 0; h < H; ++h) {
      int nf = nqt * (HD / 32);
      pack_a_frag<<<packBlocks(nf), 256, 0, stream>>>(
          qbuf + (size_t)b * T * C + h * HD, C, T, HD,
          Qp + (size_t)(b * H + h) * nqt * 8 * 512);
    }
    { int nf = nqt * (HD / 32);
      pack_b_t<<<packBlocks(nf), 256, 0, stream>>>(
          kbuf + (size_t)b * T * HD, T, HD, Kp + (size_t)b * nqt * 8 * 512); }
    { int nf = nqt * (HD / 16);
      pack_b_n<<<packBlocks(nf), 256, 0, stream>>>(
          vbuf + (size_t)b * T * HD, T, HD, Vp + (size_t)b * nqt * 16 * 512); }
  }
  // 6) attention (writes attn output into qbuf; same row-major layout)
  attn_swa_wmma<<<(B * H * nqt) / 8, 256, 0, stream>>>(
      (const v16bf*)Qp, (const v16bf*)Kp, (const v16bf*)Vp, qbuf, T);
  // 7) repack attention output (Xp is dead by now, reuse it)
  { int nf = (R / 16) * (C / 32);
    pack_a_frag<<<packBlocks(nf), 256, 0, stream>>>(qbuf, C, R, C, Xp); }
  // 8) output projection into d_out
  gemm_wmma_bf16<<<dim3(C / 64, R / 128), 256, 0, stream>>>(
      (const v16bf*)Xp, (const v16bf*)Wop, out, R, C, C);
}
